// GATNet_IMG_47528108098210
// MI455X (gfx1250) — matrix operand order repaired and checked by reading, not compiled, not run on hardware
//
#include <hip/hip_runtime.h>
#include <hip/hip_bf16.h>

// ---------------- problem constants ----------------
#define NN    4096   // nodes
#define NF    4096   // input features
#define NH    1024   // hidden per head
#define HEADS 4
#define BITS  64
#define LRA   0.2f

typedef __attribute__((ext_vector_type(16))) _Float16 v16h;
typedef __attribute__((ext_vector_type(8)))  float    v8f;

union Frag16 { v16h v; unsigned u[8]; uint4 q[2]; };

__device__ __forceinline__ unsigned pack2h(_Float16 lo, _Float16 hi) {
    union { _Float16 h[2]; unsigned u; } p;
    p.h[0] = lo; p.h[1] = hi;
    return p.u;
}

__device__ __forceinline__ v8f wmma_f32_f16(v16h a, v16h b, v8f c) {
    // D = A(16x32 f16) * B(32x16 f16) + C(16x16 f32)
    return __builtin_amdgcn_wmma_f32_16x16x32_f16(
        false, a, false, b, (short)0, c, false, false);
}

// --- CDNA5 async global->LDS copies (ASYNCcnt-tracked, §08) ---
__device__ __forceinline__ unsigned lds_addr(const void* p) {
    return (unsigned)(size_t)p;   // low 32 bits of generic ptr = LDS byte addr
}
__device__ __forceinline__ void async_copy_b128(unsigned lds, unsigned long long ga) {
    asm volatile("global_load_async_to_lds_b128 %0, %1, off" :: "v"(lds), "v"(ga) : "memory");
}
__device__ __forceinline__ void async_copy_b32(unsigned lds, unsigned long long ga) {
    asm volatile("global_load_async_to_lds_b32 %0, %1, off" :: "v"(lds), "v"(ga) : "memory");
}

// ---------------- fp32 -> f16 convert (flat, A operands) ----------------
__global__ void cvt_f32_f16(const float* __restrict__ in, _Float16* __restrict__ out, long n) {
    long i = (long)blockIdx.x * blockDim.x + threadIdx.x;
    if (i < n) out[i] = (_Float16)in[i];
}

// ------- fp32 -> f16 K-pair-packed convert (B operands) -------
// out dword (p, n) = pack(in[2p][n], in[2p+1][n]); ldn = 1<<shift columns.
__global__ void cvt_pack(const float* __restrict__ in, unsigned* __restrict__ out,
                         long strideIn, long strideOut, int shift, long total) {
    long i = (long)blockIdx.x * blockDim.x + threadIdx.x;
    if (i >= total) return;
    const float* ip = in + (long)blockIdx.z * strideIn;
    long n = i & ((1L << shift) - 1);
    long p = i >> shift;
    long base = (p << (shift + 1)) + n;
    float lo = ip[base];
    float hi = ip[base + (1L << shift)];
    out[(long)blockIdx.z * strideOut + i] = pack2h((_Float16)lo, (_Float16)hi);
}

// ---------------- tiled WMMA GEMM: C(f32) = A(f16,MxK) * Bpk ------------
// block = 256 threads = 8 waves; tile 128x64, K-step 32; double-buffered
// async staging. Bpk is the K-pair-packed B (dword (p,n)).
__global__ __launch_bounds__(256)
void gemm_f16_wmma(const _Float16* __restrict__ A, long strideA, int lda,
                   const unsigned* __restrict__ Bpk, long strideB, int ldnB,
                   float* __restrict__ C, long strideC, int ldc, int K) {
    A   += (long)blockIdx.z * strideA;
    Bpk += (long)blockIdx.z * strideB;
    C   += (long)blockIdx.z * strideC;

    const int m0 = blockIdx.y * 128;
    const int n0 = blockIdx.x * 64;
    const int tid  = threadIdx.x;
    const int wave = tid >> 5;           // 0..7 -> 16-row slice
    const int lane = tid & 31;
    const int mL = lane & 15;
    const int g  = lane >> 4;
    const int nL = lane & 15;

    __shared__ _Float16 sA[2][128][40];  // 80B rows: aligned + conflict-free
    __shared__ unsigned sB[2][64][20];   // col-major packed pairs, 80B rows

    auto issueTile = [&](int k0, int buf) {
        // A tile 128x32 f16: 2 async b128 per thread
        #pragma unroll
        for (int i = 0; i < 2; ++i) {
            int slot = i * 256 + tid;
            int row = slot >> 2, seg = slot & 3;
            async_copy_b128(lds_addr(&sA[buf][row][seg * 8]),
                (unsigned long long)(size_t)(A + (long)(m0 + row) * lda + k0 + seg * 8));
        }
        // B tile: 16 pair-rows x 64 cols -> transposed scatter, 4 async b32
        int kp = k0 >> 1;
        #pragma unroll
        for (int i = 0; i < 4; ++i) {
            int d = i * 256 + tid;
            int p = d >> 6, n = d & 63;
            async_copy_b32(lds_addr(&sB[buf][n][p]),
                (unsigned long long)(size_t)(Bpk + (long)(kp + p) * ldnB + n0 + n));
        }
    };

    v8f acc[4] = {{}, {}, {}, {}};
    const int nsteps = K >> 5;

    issueTile(0, 0);
    for (int s = 0; s < nsteps; ++s) {
        const int buf = s & 1;
        const bool hasNext = (s + 1) < nsteps;
        if (hasNext) issueTile((s + 1) << 5, buf ^ 1);
        if (hasNext) asm volatile("s_wait_asynccnt 0x6" ::: "memory");
        else         asm volatile("s_wait_asynccnt 0x0" ::: "memory");
        __syncthreads();

        // batch ALL fragment loads first, then stream 4 independent WMMAs
        Frag16 af, bf[4];
        const unsigned* rowp = (const unsigned*)&sA[buf][wave * 16 + mL][0];
        af.q[0] = *(const uint4*)&rowp[4 * g];
        af.q[1] = *(const uint4*)&rowp[8 + 4 * g];
        #pragma unroll
        for (int c = 0; c < 4; ++c) {
            const unsigned* colp = &sB[buf][c * 16 + nL][0];
            bf[c].q[0] = *(const uint4*)&colp[4 * g];
            bf[c].q[1] = *(const uint4*)&colp[8 + 4 * g];
        }
        #pragma unroll
        for (int c = 0; c < 4; ++c)
            acc[c] = wmma_f32_f16(af.v, bf[c].v, acc[c]);
        __syncthreads();
    }

    #pragma unroll
    for (int c = 0; c < 4; ++c)
        #pragma unroll
        for (int r = 0; r < 8; ++r) {
            long row = m0 + wave * 16 + 8 * g + r;
            long col = n0 + c * 16 + nL;
            C[row * ldc + col] = acc[c][r];
        }
}

// ---------------- f1/f2 = Wh . a1 / a2 per (head,row) ----------------
__global__ __launch_bounds__(256)
void f1f2_kernel(const float* __restrict__ Wh, const float* __restrict__ a1,
                 const float* __restrict__ a2, float* __restrict__ f1, float* __restrict__ f2) {
    const int h = blockIdx.y, n = blockIdx.x, t = threadIdx.x;
    const long base = ((long)h * NN + n) * NH;
    const float* a1p = a1 + h * NH;
    const float* a2p = a2 + h * NH;
    float s1 = 0.f, s2 = 0.f;
    for (int c = t; c < NH; c += 256) {
        float w = Wh[base + c];
        s1 += w * a1p[c]; s2 += w * a2p[c];
    }
    __shared__ float r1[256], r2[256];
    r1[t] = s1; r2[t] = s2; __syncthreads();
    for (int off = 128; off > 0; off >>= 1) {
        if (t < off) { r1[t] += r1[t + off]; r2[t] += r2[t + off]; }
        __syncthreads();
    }
    if (t == 0) { f1[(long)h * NN + n] = r1[0]; f2[(long)h * NN + n] = r2[0]; }
}

// ------------- per-row masked softmax stats: rowmax / rowsum -------------
__global__ __launch_bounds__(256)
void rowstats_kernel(const float* __restrict__ f1, const float* __restrict__ f2,
                     const float* __restrict__ adj, float* __restrict__ rmax,
                     float* __restrict__ rsum) {
    const int h = blockIdx.y, i = blockIdx.x, t = threadIdx.x;
    const float f1v = f1[(long)h * NN + i];
    const float* f2p = f2 + (long)h * NN;
    const float* ar = adj + (long)i * NN;
    __shared__ float sred[256];

    float m = -3.0e38f;
    for (int j = t; j < NN; j += 256) {
        if (ar[j] > 0.f) {
            float z = f1v + f2p[j];
            float e = z > 0.f ? z : LRA * z;
            m = fmaxf(m, e);
        }
    }
    sred[t] = m; __syncthreads();
    for (int off = 128; off > 0; off >>= 1) {
        if (t < off) sred[t] = fmaxf(sred[t], sred[t + off]);
        __syncthreads();
    }
    const float rowm = sred[0];
    __syncthreads();

    float s = 0.f;
    if (rowm > -2.9e38f) {
        for (int j = t; j < NN; j += 256) {
            if (ar[j] > 0.f) {
                float z = f1v + f2p[j];
                float e = z > 0.f ? z : LRA * z;
                s += __expf(e - rowm);
            }
        }
    }
    sred[t] = s; __syncthreads();
    for (int off = 128; off > 0; off >>= 1) {
        if (t < off) sred[t] += sred[t + off];
        __syncthreads();
    }
    if (t == 0) { rmax[(long)h * NN + i] = rowm; rsum[(long)h * NN + i] = sred[0]; }
}

// ------ g1/g2 for output layer (dot over BITS=64) ------
__global__ void gvec_kernel(const float* __restrict__ Wh2, const float* __restrict__ a1o,
                            const float* __restrict__ a2o, float* __restrict__ g1,
                            float* __restrict__ g2) {
    int n = blockIdx.x * blockDim.x + threadIdx.x;
    if (n >= NN) return;
    float s1 = 0.f, s2 = 0.f;
    const float* row = Wh2 + (long)n * BITS;
    #pragma unroll
    for (int b = 0; b < BITS; ++b) { s1 += row[b] * a1o[b]; s2 += row[b] * a2o[b]; }
    g1[n] = s1; g2[n] = s2;
}

// --------- fused masked-softmax attention aggregation via WMMA ----------
// Softmax-prob A fragment generated in registers (ISA 16-bit A layout);
// B comes pair-packed via double-buffered async LDS tiles.
// MODE 0: scale + ELU + f16 store into x_cat.  MODE 1: scale + tanh + f32 out.
// CPW = column tiles per wave (block covers 64*CPW output columns).
template <int MODE, int CPW>
__global__ __launch_bounds__(128)
void agg_kernel(const float* __restrict__ f1, const float* __restrict__ f2,
                const float* __restrict__ rmax, const float* __restrict__ rsum,
                const float* __restrict__ adj,
                const unsigned* __restrict__ Bpk, int ldnB, long strideB,
                void* __restrict__ outv, int outld, int ncols) {
    constexpr int COLS = 64 * CPW;
    constexpr int CSH  = (CPW == 2) ? 7 : 6;

    const int h  = blockIdx.z;
    const int i0 = blockIdx.y * 16;
    const int o0 = blockIdx.x * COLS;
    f1 += (long)h * NN; f2 += (long)h * NN;
    rmax += (long)h * NN; rsum += (long)h * NN;
    Bpk += (long)h * strideB;

    const int tid  = threadIdx.x;
    const int wave = tid >> 5;           // 0..3
    const int lane = tid & 31;
    const int mL = lane & 15;
    const int g  = lane >> 4;
    const int nL = lane & 15;

    __shared__ float    sF1[16], sRM[16], sSc[16], sUni[16];
    __shared__ float    sF2[2][32];
    __shared__ float    sAdj[2][16][36];   // 144B rows: aligned + conflict-free
    __shared__ unsigned sB[2][COLS][20];   // col-major packed pairs

    if (tid < 16) {
        int m = tid;
        sF1[m] = f1[i0 + m];
        sRM[m] = rmax[i0 + m];
        float s = rsum[i0 + m];
        bool uni = !(s > 0.f);
        sUni[m] = uni ? 1.f : 0.f;
        sSc[m]  = uni ? (1.0f / (float)NN) : (1.0f / s);
    }

    auto issueTile = [&](int j0, int buf) {
        if (tid < 32) sF2[buf][tid] = f2[j0 + tid];
        { // adj tile 16x32 f32: 1 async b128 per thread
            int m = tid >> 3, oc = tid & 7;
            async_copy_b128(lds_addr(&sAdj[buf][m][oc * 4]),
                (unsigned long long)(size_t)(adj + (long)(i0 + m) * NN + j0 + oc * 4));
        }
        { // B tile: 16 pair-rows x COLS cols -> transposed scatter
            int kp = j0 >> 1;
            #pragma unroll
            for (int i = 0; i < 8 * CPW; ++i) {
                int d = i * 128 + tid;
                int p = d >> CSH, n = d & (COLS - 1);
                async_copy_b32(lds_addr(&sB[buf][n][p]),
                    (unsigned long long)(size_t)(Bpk + (long)(kp + p) * ldnB + o0 + n));
            }
        }
    };

    v8f acc[CPW];
    #pragma unroll
    for (int c = 0; c < CPW; ++c) acc[c] = {};

    const int nsteps = NN / 32;
    issueTile(0, 0);
    for (int s = 0; s < nsteps; ++s) {
        const int buf = s & 1;
        const bool hasNext = (s + 1) < nsteps;
        if (hasNext) issueTile((s + 1) << 5, buf ^ 1);
        if (hasNext) {
            if constexpr (CPW == 2) asm volatile("s_wait_asynccnt 0x11" ::: "memory");
            else                    asm volatile("s_wait_asynccnt 0x9" ::: "memory");
        } else {
            asm volatile("s_wait_asynccnt 0x0" ::: "memory");
        }
        __syncthreads();

        // load B fragments first, then generate A fragment, then stream WMMAs
        Frag16 bf[CPW];
        #pragma unroll
        for (int c = 0; c < CPW; ++c) {
            const int coloff = (wave + 4 * c) * 16;
            const unsigned* colp = &sB[buf][coloff + nL][0];
            bf[c].q[0] = *(const uint4*)&colp[4 * g];
            bf[c].q[1] = *(const uint4*)&colp[8 + 4 * g];
        }

        // generate attention-prob A fragment in-register (ISA 16-bit A layout)
        const float f1v = sF1[mL], rm = sRM[mL];
        const bool  uni = sUni[mL] > 0.5f;
        Frag16 af;
        #pragma unroll
        for (int v = 0; v < 8; ++v) {
            int kb = 2 * (v & 3) + 16 * (v >> 2) + 8 * g;
            _Float16 hv[2];
            #pragma unroll
            for (int e = 0; e < 2; ++e) {
                int k = kb + e;
                float z = f1v + sF2[buf][k];
                z = z > 0.f ? z : LRA * z;
                float p;
                if (uni)                           p = 1.0f;
                else if (sAdj[buf][mL][k] > 0.f)   p = __expf(z - rm);
                else                               p = 0.0f;
                hv[e] = (_Float16)p;
            }
            af.u[v] = pack2h(hv[0], hv[1]);
        }

        #pragma unroll
        for (int c = 0; c < CPW; ++c)
            acc[c] = wmma_f32_f16(af.v, bf[c].v, acc[c]);
        __syncthreads();
    }

    #pragma unroll
    for (int c = 0; c < CPW; ++c)
        #pragma unroll
        for (int r = 0; r < 8; ++r) {
            int mloc = 8 * g + r;
            long row = i0 + mloc;
            long col = (long)blockIdx.z * ncols + o0 + (wave + 4 * c) * 16 + nL;
            float v = acc[c][r] * sSc[mloc];
            if (MODE == 0) {
                v = v > 0.f ? v : (__expf(v) - 1.0f);            // ELU
                ((_Float16*)outv)[row * outld + col] = (_Float16)v;
            } else {
                ((float*)outv)[row * outld + col] = tanhf(v);    // alp=1.0
            }
        }
}

// =====================================================================
extern "C" void kernel_launch(void* const* d_in, const int* in_sizes, int n_in,
                              void* d_out, int out_size, void* d_ws, size_t ws_size,
                              hipStream_t stream) {
    const float* x    = (const float*)d_in[0];   // [NN,NF]
    const float* adj  = (const float*)d_in[1];   // [NN,NN]
    const float* W    = (const float*)d_in[2];   // [H,NF,NH]
    const float* a1   = (const float*)d_in[3];   // [H,NH]
    const float* a2   = (const float*)d_in[4];   // [H,NH]
    const float* Wout = (const float*)d_in[5];   // [H*NH,BITS]
    const float* a1o  = (const float*)d_in[6];   // [BITS]
    const float* a2o  = (const float*)d_in[7];   // [BITS]
    float* out = (float*)d_out;                  // [NN,BITS]

    const long szX    = (long)NN * NF;
    const long szW    = (long)HEADS * NF * NH;
    const long szWh   = (long)HEADS * NN * NH;
    const long szXcat = (long)NN * HEADS * NH;
    const long szWout = (long)HEADS * NH * BITS;
    const long szWh2  = (long)NN * BITS;

    char* cur = (char*)d_ws;
    auto carve = [&](size_t bytes) {
        void* p = cur;
        cur += (bytes + 255) & ~(size_t)255;
        return p;
    };
    _Float16* xh      = (_Float16*)carve(szX * 2);
    unsigned* w16p    = (unsigned*)carve(szW * 2);     // pair-packed
    float*    whF     = (float*)   carve(szWh * 4);
    unsigned* wh16p   = (unsigned*)carve(szWh * 2);    // pair-packed
    _Float16* xcat16  = (_Float16*)carve(szXcat * 2);
    unsigned* wout16p = (unsigned*)carve(szWout * 2);  // pair-packed
    float*    wh2F    = (float*)   carve(szWh2 * 4);
    unsigned* wh2hp   = (unsigned*)carve(szWh2 * 2);   // pair-packed
    float* f1    = (float*)carve((long)HEADS * NN * 4);
    float* f2    = (float*)carve((long)HEADS * NN * 4);
    float* rmax_ = (float*)carve((long)HEADS * NN * 4);
    float* rsum_ = (float*)carve((long)HEADS * NN * 4);
    float* g1    = (float*)carve((long)NN * 4);
    float* g2    = (float*)carve((long)NN * 4);
    float* rmax2 = (float*)carve((long)NN * 4);
    float* rsum2 = (float*)carve((long)NN * 4);

    // --- operand conversions: A flat f16, B K-pair-packed f16 ---
    cvt_f32_f16<<<(unsigned)((szX + 255) / 256), 256, 0, stream>>>(x, xh, szX);
    {
        long tot = (long)(NF / 2) * NH;   // per head
        cvt_pack<<<dim3((unsigned)((tot + 255) / 256), 1, HEADS), 256, 0, stream>>>(
            W, w16p, (long)NF * NH, tot, 10, tot);
    }
    {
        long tot = (long)(HEADS * NH / 2) * BITS;
        cvt_pack<<<dim3((unsigned)((tot + 255) / 256), 1, 1), 256, 0, stream>>>(
            Wout, wout16p, 0L, 0L, 6, tot);
    }

    // --- Wh[h] = x @ W[h]  (batched over heads) ---
    gemm_f16_wmma<<<dim3(NH / 64, NN / 128, HEADS), 256, 0, stream>>>(
        xh, 0L, NF, w16p, (long)(NF / 2) * NH, NH,
        whF, (long)NN * NH, NH, NF);

    // --- f1/f2 and pair-packed Wh for the aggregation B operand ---
    f1f2_kernel<<<dim3(NN, HEADS), 256, 0, stream>>>(whF, a1, a2, f1, f2);
    {
        long tot = (long)(NN / 2) * NH;   // per head
        cvt_pack<<<dim3((unsigned)((tot + 255) / 256), 1, HEADS), 256, 0, stream>>>(
            whF, wh16p, (long)NN * NH, tot, 10, tot);
    }

    // --- masked-softmax row stats ---
    rowstats_kernel<<<dim3(NN, HEADS), 256, 0, stream>>>(f1, f2, adj, rmax_, rsum_);

    // --- fused attention aggregation + ELU -> x_cat (f16) ---
    agg_kernel<0, 2><<<dim3(NH / 128, NN / 16, HEADS), 128, 0, stream>>>(
        f1, f2, rmax_, rsum_, adj, wh16p, NH, (long)(NN / 2) * NH,
        (void*)xcat16, HEADS * NH, NH);

    // --- Wh2 = x_cat @ W_out ---
    gemm_f16_wmma<<<dim3(BITS / 64, NN / 128, 1), 256, 0, stream>>>(
        xcat16, 0L, HEADS * NH, wout16p, 0L, BITS,
        wh2F, 0L, BITS, HEADS * NH);

    // --- output attention layer ---
    gvec_kernel<<<(NN + 255) / 256, 256, 0, stream>>>(wh2F, a1o, a2o, g1, g2);
    {
        long tot = (long)(NN / 2) * BITS;
        cvt_pack<<<dim3((unsigned)((tot + 255) / 256), 1, 1), 256, 0, stream>>>(
            wh2F, wh2hp, 0L, 0L, 6, tot);
    }
    rowstats_kernel<<<dim3(NN, 1), 256, 0, stream>>>(g1, g2, adj, rmax2, rsum2);

    // --- fused aggregation + tanh -> d_out (f32) ---
    agg_kernel<1, 1><<<dim3(1, NN / 16, 1), 128, 0, stream>>>(
        g1, g2, rmax2, rsum2, adj, wh2hp, BITS, 0L,
        (void*)out, BITS, BITS);

    (void)in_sizes; (void)n_in; (void)out_size; (void)ws_size;
}
